// STALSTM_42142219108667
// MI455X (gfx1250) — matrix operand order repaired and checked by reading, not compiled, not run on hardware
//
#include <hip/hip_runtime.h>
#include <hip/hip_bf16.h>
#include <stdint.h>

// ---------------- problem constants ----------------
#define T_  512
#define B_  128
#define J_  25
#define JP_ 32          // J padded to 32 for WMMA K-tiling
#define H_  512
#define G4_ 2048        // 4*H
#define K0_ (H_ + JP_)      // 544 : recurrent K for in_dim=25 layers
#define K12_ (H_ + 2*H_)    // 1536: recurrent K for in_dim=1024 layers

typedef __attribute__((ext_vector_type(16))) __bf16 v16bf;
typedef __attribute__((ext_vector_type(8)))  float  v8f;

union V16 {
    v16bf v;
    uint4 u4[2];
};

__device__ __forceinline__ unsigned short f2bf(float f) {
    unsigned u = __float_as_uint(f);
    unsigned r = (u + 0x7FFFu + ((u >> 16) & 1u)) >> 16;   // RNE
    return (unsigned short)r;
}
__device__ __forceinline__ float sigmoidf_(float x) {
    return 1.0f / (1.0f + __expf(-x));
}

// ---------------- weight / input conversion ----------------

// Wcat[g][k] (bf16, row-major, ld=Kcat): k<H -> Whh[g][k]; else Wih[g][k-H] (zero pad)
__global__ void k_cvt_wcat(unsigned short* __restrict__ Wc,
                           const float* __restrict__ Whh,
                           const float* __restrict__ Wih,
                           int in_dim, int Kcat) {
    size_t idx = (size_t)blockIdx.x * blockDim.x + threadIdx.x;
    size_t tot = (size_t)G4_ * Kcat;
    if (idx >= tot) return;
    int k = (int)(idx % Kcat);
    int g = (int)(idx / Kcat);
    float v;
    if (k < H_) {
        v = Whh[(size_t)g * H_ + k];
    } else {
        int kk = k - H_;
        v = (kk < in_dim) ? Wih[(size_t)g * in_dim + kk] : 0.0f;
    }
    Wc[idx] = f2bf(v);
}

__global__ void k_bias_combine(float* __restrict__ bout,
                               const float* __restrict__ bih,
                               const float* __restrict__ bhh) {
    int i = blockIdx.x * blockDim.x + threadIdx.x;
    if (i < G4_) bout[i] = bih[i] + bhh[i];
}

// xpad[t][b][j] = bf16(x[t][b][j]) for j<25, else 0
__global__ void k_cvt_xpad(unsigned short* __restrict__ xpad,
                           const float* __restrict__ x) {
    size_t idx = (size_t)blockIdx.x * blockDim.x + threadIdx.x;   // T*B*32
    if (idx >= (size_t)T_ * B_ * JP_) return;
    int j = (int)(idx & (JP_ - 1));
    size_t tb = idx >> 5;
    float v = (j < J_) ? x[tb * J_ + j] : 0.0f;
    xpad[idx] = f2bf(v);
}

// y0pad[t][b][j] = bf16(x[t][b][j] * alpha[b][j])
__global__ void k_apply_alpha(unsigned short* __restrict__ ypad,
                              const float* __restrict__ x,
                              const float* __restrict__ alpha) {
    size_t idx = (size_t)blockIdx.x * blockDim.x + threadIdx.x;
    if (idx >= (size_t)T_ * B_ * JP_) return;
    int j = (int)(idx & (JP_ - 1));
    size_t tb = idx >> 5;
    int b = (int)(tb % B_);
    float v = 0.0f;
    if (j < J_) v = x[tb * J_ + j] * alpha[b * J_ + j];
    ypad[idx] = f2bf(v);
}

__global__ void k_init_state(unsigned short* __restrict__ hbf,
                             float* __restrict__ cbuf,
                             float* __restrict__ hf32,
                             const float* __restrict__ h0,
                             const float* __restrict__ c0) {
    int idx = blockIdx.x * blockDim.x + threadIdx.x;  // 2*B*H
    if (idx >= 2 * B_ * H_) return;
    float h = h0[idx];
    hbf[idx] = f2bf(h);
    hf32[idx] = h;
    cbuf[idx] = c0[idx];
}

// ---------------- WMMA recurrent step GEMM ----------------
// z[dir][m][n] = bias[dir][n] + sum_k A[m,k]*Wcat[n,k]
//   A row m: k<H -> hbf[dir][m][k];  k>=H -> xsrc[row(dir,t)][m][k-H]
// grid: 256 blocks x 256 threads. block = 8 waves; wave w -> mtile w.
// blockIdx.x: low 7 bits = ntile (0..127), bit7 = dir.
__global__ void k_lstm_gemm_step(float* __restrict__ z,             // [2][B][G4]
                                 const unsigned short* __restrict__ Wc0,
                                 const unsigned short* __restrict__ Wc1,
                                 const float* __restrict__ bias,    // [2][G4]
                                 const unsigned short* __restrict__ hbf, // [2][B][H]
                                 const unsigned short* __restrict__ xsrc,// [T][B][xdim]
                                 int xdim, int Kcat, int t) {
    const int wave = threadIdx.x >> 5;     // 0..7 -> mtile
    const int lane = threadIdx.x & 31;
    const int ntile = blockIdx.x & 127;
    const int dir = blockIdx.x >> 7;

    const unsigned short* W = dir ? Wc1 : Wc0;
    const int mBase = wave * 16;
    const int nBase = ntile * 16;
    const int half = lane >> 4;            // 0 or 1
    const int l15 = lane & 15;

    const int aRow = mBase + l15;          // m index this lane loads for A
    const int bRow = nBase + l15;          // n index this lane loads for B
    const int xrow = dir ? (T_ - 1 - t) : t;

    const unsigned short* hA = hbf + (size_t)dir * B_ * H_ + (size_t)aRow * H_;
    const unsigned short* xA = xsrc + ((size_t)xrow * B_ + aRow) * xdim - H_;
    const unsigned short* Wr = W + (size_t)bRow * Kcat;

    const int aOff = half * 8;             // bf16 A 16x32 lane layout (ISA 7.12.2)
    const int bOff = half * 8;             // bf16 B 32x16 lane layout

    v8f acc = {};
    for (int kb = 0; kb < Kcat; kb += 32) {
        V16 a, b;
        const unsigned short* ap = ((kb < H_) ? hA : xA) + kb + aOff;
        a.u4[0] = *(const uint4*)(ap);
        a.u4[1] = *(const uint4*)(ap + 16);
        const unsigned short* bp = Wr + kb + bOff;
        __builtin_prefetch(bp + 64, 0, 1);   // global_prefetch_b8: prime next K tiles
        b.u4[0] = *(const uint4*)(bp);
        b.u4[1] = *(const uint4*)(bp + 16);
        acc = __builtin_amdgcn_wmma_f32_16x16x32_bf16(
                  false, a.v, false, b.v, (short)0, acc, false, false);
    }

    const float bn = bias[dir * G4_ + nBase + l15];
    float* zo = z + (size_t)dir * B_ * G4_;
#pragma unroll
    for (int r = 0; r < 8; ++r) {
        int row = mBase + r + half * 8;    // D layout: lanes 0-15 M=r, 16-31 M=r+8
        zo[(size_t)row * G4_ + nBase + l15] = acc[r] + bn;
    }
}

// ---------------- LSTM cell pointwise step ----------------
__global__ void k_lstm_cell_step(const float* __restrict__ z,   // [2][B][G4]
                                 float* __restrict__ cbuf,      // [2][B][H]
                                 unsigned short* __restrict__ hbf,
                                 float* __restrict__ hf32,
                                 unsigned short* __restrict__ yout, // [T][B][2H] or null
                                 int t) {
    int idx = blockIdx.x * blockDim.x + threadIdx.x;  // 2*B*H = 131072
    if (idx >= 2 * B_ * H_) return;
    int dir = idx >> 16;
    int r = idx & 65535;
    int m = r >> 9;
    int n = r & 511;
    const float* zd = z + (size_t)dir * B_ * G4_ + (size_t)m * G4_;
    float zi = zd[n];
    float zf = zd[n + 512];
    float zg = zd[n + 1024];
    float zo = zd[n + 1536];
    float c = cbuf[idx];
    c = sigmoidf_(zf) * c + sigmoidf_(zi) * tanhf(zg);
    float h = sigmoidf_(zo) * tanhf(c);
    cbuf[idx] = c;
    hbf[idx] = f2bf(h);
    hf32[idx] = h;
    if (yout) {
        int row = dir ? (T_ - 1 - t) : t;
        yout[((size_t)row * B_ + m) * (2 * H_) + dir * H_ + n] = f2bf(h);
    }
}

// ---------------- small FC: out[m][n] = act(in[m,:] . W[n,:] + b[n]) ----------------
__global__ void k_fc(const float* __restrict__ in, const float* __restrict__ W,
                     const float* __restrict__ bias, float* __restrict__ out,
                     int M, int N, int K, int act) {
    int idx = blockIdx.x * blockDim.x + threadIdx.x;
    if (idx >= M * N) return;
    int m = idx / N, n = idx - m * N;
    const float* a = in + (size_t)m * K;
    const float* w = W + (size_t)n * K;
    float s = bias[n];
    for (int k = 0; k < K; ++k) s += a[k] * w[k];
    if (act == 1) s = tanhf(s);
    else if (act == 2) s = __expf(s);
    out[idx] = s;
}

// ---------------- attention normalize: alpha[b][j] = sum_dir a / sum_{dir,j} a --------
__global__ void k_attn_norm(const float* __restrict__ a,  // [2][B][J]
                            float* __restrict__ alpha) {  // [B][J]
    int b = blockIdx.x;
    int lane = threadIdx.x;   // 32 threads
    float s = 0.0f;
    for (int i = lane; i < 2 * J_; i += 32) {
        int dir = i / J_, j = i - dir * J_;
        s += a[((size_t)dir * B_ + b) * J_ + j];
    }
    for (int off = 16; off; off >>= 1) s += __shfl_down(s, off, 32);
    s = __shfl(s, 0, 32);
    for (int j = lane; j < J_; j += 32)
        alpha[b * J_ + j] = (a[(size_t)b * J_ + j] + a[((size_t)B_ + b) * J_ + j]) / s;
}

// ---------------- body-part value (threshold + segment-sum + normalize) ----------------
__global__ void k_bpv(const float* __restrict__ alpha,
                      const int* __restrict__ part_idx,
                      float* __restrict__ outtail) {       // 5 floats
    __shared__ float parts[5];
    int lane = threadIdx.x;
    if (lane < 5) parts[lane] = 0.0f;
    __syncthreads();
    if (lane < J_) {
        float s = 0.0f;
        for (int b = 0; b < B_; ++b) {
            float v = alpha[b * J_ + lane];
            if (v >= 0.1f) s += v;
        }
        atomicAdd(&parts[part_idx[lane]], s);
    }
    __syncthreads();
    if (lane == 0) {
        float tot = parts[0] + parts[1] + parts[2] + parts[3] + parts[4] + 1e-8f;
        for (int p = 0; p < 5; ++p) outtail[p] = parts[p] / tot;
    }
}

// ---------------- host orchestration ----------------
extern "C" void kernel_launch(void* const* d_in, const int* in_sizes, int n_in,
                              void* d_out, int out_size, void* d_ws, size_t ws_size,
                              hipStream_t stream) {
    if (n_in < 44) return;
    const float* x        = (const float*)d_in[0];
    const float* h0_sa    = (const float*)d_in[1];
    const float* c0_sa    = (const float*)d_in[2];
    const float* h0_main  = (const float*)d_in[3];
    const float* c0_main  = (const float*)d_in[4];
    // sa params: [5..12] = fwd(Wih,Whh,bih,bhh), bwd(Wih,Whh,bih,bhh)
    // main params: base 13, layer l dir d: 13 + 8*l + 4*d + {0:Wih,1:Whh,2:bih,3:bhh}
    const float* safc1_w  = (const float*)d_in[37];
    const float* safc1_b  = (const float*)d_in[38];
    const float* safc2_w  = (const float*)d_in[39];
    const float* safc2_b  = (const float*)d_in[40];
    const float* mfc1_w   = (const float*)d_in[41];
    const float* mfc1_b   = (const float*)d_in[42];
    const int*   part_idx = (const int*)d_in[43];
    float* out = (float*)d_out;

    // --- bump allocator over workspace (256B aligned) ---
    char* base = (char*)d_ws;
    size_t off = 0;
    auto alloc = [&](size_t bytes) -> char* {
        char* p = base + off;
        off = (off + bytes + 255) & ~(size_t)255;
        return p;
    };
    // bf16 weight concat buffers: slots 0,1=sa; 2,3=main l0 (Kcat=544); 4..7=main l1,l2 (1536)
    unsigned short* Wc[8];
    for (int s = 0; s < 4; ++s) Wc[s] = (unsigned short*)alloc((size_t)G4_ * K0_ * 2);
    for (int s = 4; s < 8; ++s) Wc[s] = (unsigned short*)alloc((size_t)G4_ * K12_ * 2);
    float* bias8 = (float*)alloc(8 * G4_ * 4);
    unsigned short* xpad  = (unsigned short*)alloc((size_t)T_ * B_ * JP_ * 2);
    unsigned short* y0pad = (unsigned short*)alloc((size_t)T_ * B_ * JP_ * 2);
    unsigned short* ybufA = (unsigned short*)alloc((size_t)T_ * B_ * 2 * H_ * 2);
    unsigned short* ybufB = (unsigned short*)alloc((size_t)T_ * B_ * 2 * H_ * 2);
    unsigned short* hbf   = (unsigned short*)alloc((size_t)2 * B_ * H_ * 2);
    float* cbuf  = (float*)alloc((size_t)2 * B_ * H_ * 4);
    float* hf32  = (float*)alloc((size_t)8 * B_ * H_ * 4);   // slots: 0,1 sa; 2..7 main
    float* zbuf  = (float*)alloc((size_t)2 * B_ * G4_ * 4);
    float* h1buf = (float*)alloc((size_t)2 * B_ * H_ * 4);
    float* abuf  = (float*)alloc((size_t)2 * B_ * J_ * 4);
    float* alpha = (float*)alloc((size_t)B_ * J_ * 4);
    (void)ws_size; (void)in_sizes; (void)out_size;

    const int TPB = 256;

    // --- convert weights/biases (slots: 0,1 sa; 2,3 l0; 4,5 l1; 6,7 l2) ---
    for (int s = 0; s < 8; ++s) {
        int pbase, in_dim, Kcat;
        if (s < 2)      { pbase = 5 + 4 * s;            in_dim = J_;     Kcat = K0_;  }
        else            { int l = (s - 2) / 2, d = s & 1;
                          pbase = 13 + 8 * l + 4 * d;
                          in_dim = (l == 0) ? J_ : 2 * H_;
                          Kcat = (l == 0) ? K0_ : K12_; }
        const float* Wih = (const float*)d_in[pbase + 0];
        const float* Whh = (const float*)d_in[pbase + 1];
        const float* bih = (const float*)d_in[pbase + 2];
        const float* bhh = (const float*)d_in[pbase + 3];
        size_t tot = (size_t)G4_ * Kcat;
        k_cvt_wcat<<<dim3((unsigned)((tot + TPB - 1) / TPB)), TPB, 0, stream>>>(
            Wc[s], Whh, Wih, in_dim, Kcat);
        k_bias_combine<<<(G4_ + TPB - 1) / TPB, TPB, 0, stream>>>(
            bias8 + (size_t)s * G4_, bih, bhh);
    }

    // --- input pad/convert ---
    {
        size_t tot = (size_t)T_ * B_ * JP_;
        k_cvt_xpad<<<dim3((unsigned)((tot + TPB - 1) / TPB)), TPB, 0, stream>>>(xpad, x);
    }

    const int cellBlocks = (2 * B_ * H_ + TPB - 1) / TPB;

    // --- SA bidirectional LSTM scan ---
    k_init_state<<<cellBlocks, TPB, 0, stream>>>(hbf, cbuf, hf32 + 0, h0_sa, c0_sa);
    for (int t = 0; t < T_; ++t) {
        k_lstm_gemm_step<<<256, 256, 0, stream>>>(zbuf, Wc[0], Wc[1], bias8 + 0,
                                                  hbf, xpad, JP_, K0_, t);
        k_lstm_cell_step<<<cellBlocks, TPB, 0, stream>>>(zbuf, cbuf, hbf, hf32 + 0,
                                                         (unsigned short*)nullptr, t);
    }

    // --- attention head ---
    k_fc<<<(2 * B_ * H_ + TPB - 1) / TPB, TPB, 0, stream>>>(
        hf32 + 0, safc1_w, safc1_b, h1buf, 2 * B_, H_, H_, /*tanh*/1);
    k_fc<<<(2 * B_ * J_ + TPB - 1) / TPB, TPB, 0, stream>>>(
        h1buf, safc2_w, safc2_b, abuf, 2 * B_, J_, H_, /*exp*/2);
    k_attn_norm<<<B_, 32, 0, stream>>>(abuf, alpha);
    k_bpv<<<1, 32, 0, stream>>>(alpha, part_idx, out + 6 * B_ * H_);
    {
        size_t tot = (size_t)T_ * B_ * JP_;
        k_apply_alpha<<<dim3((unsigned)((tot + TPB - 1) / TPB)), TPB, 0, stream>>>(
            y0pad, x, alpha);
    }

    // --- main 3-layer bidirectional LSTM ---
    const unsigned short* layerIn = y0pad;
    int xdim = JP_, Kcat = K0_;
    for (int l = 0; l < 3; ++l) {
        unsigned short* yout = (l == 0) ? ybufA : (l == 1) ? ybufB : (unsigned short*)nullptr;
        float* hslot = hf32 + (size_t)(2 + 2 * l) * B_ * H_;
        k_init_state<<<cellBlocks, TPB, 0, stream>>>(
            hbf, cbuf, hslot, h0_main + (size_t)2 * l * B_ * H_,
            c0_main + (size_t)2 * l * B_ * H_);
        for (int t = 0; t < T_; ++t) {
            k_lstm_gemm_step<<<256, 256, 0, stream>>>(
                zbuf, Wc[2 + 2 * l], Wc[3 + 2 * l], bias8 + (size_t)(2 + 2 * l) * G4_,
                hbf, layerIn, xdim, Kcat, t);
            k_lstm_cell_step<<<cellBlocks, TPB, 0, stream>>>(
                zbuf, cbuf, hbf, hslot, yout, t);
        }
        layerIn = yout;
        xdim = 2 * H_;
        Kcat = K12_;
    }

    // --- final FC: out[6,B,H] = hn @ mainfc1_w^T + b ---
    k_fc<<<(6 * B_ * H_ + TPB - 1) / TPB, TPB, 0, stream>>>(
        hf32 + (size_t)2 * B_ * H_, mfc1_w, mfc1_b, out, 6 * B_, H_, H_, /*none*/0);
}
